// VisionTransformer_48206712930908
// MI455X (gfx1250) — compile-verified, matrix operand order checked by reading, not run on hardware
//
#include <hip/hip_runtime.h>
#include <hip/hip_bf16.h>
#include <math.h>
#include <stdint.h>

typedef __attribute__((ext_vector_type(16))) _Float16 v16h;
typedef __attribute__((ext_vector_type(8)))  _Float16 v8h;
typedef __attribute__((ext_vector_type(4)))  _Float16 v4h;
typedef __attribute__((ext_vector_type(8)))  float    v8f;
typedef __attribute__((ext_vector_type(4)))  float    v4f;

// ---------------------------------------------------------------------------
// WMMA fragment index helper (CDNA5 ISA 7.12.2, 16-bit A 16x32 layout):
// lane l<16: row M=l, K={0..7,16..23}; lane l+16: row M=l, K={8..15,24..31}.
// element j (0..15):  k = ((j>>3)<<4) + ((lane>>4)<<3) + (j&7)
// B (32x16) mirrors with N across lanes. C/D f32: row = 8*(lane>>4)+i, col = lane&15.
// ---------------------------------------------------------------------------
__device__ __forceinline__ int frag_k(int lane, int j) {
    return ((j >> 3) << 4) + ((lane >> 4) << 3) + (j & 7);
}

// Async copy 16 bytes global -> LDS (CDNA5 GLOBAL_LOAD_ASYNC_TO_LDS_B128,
// tracked by ASYNCcnt). ldsaddr is the 32-bit LDS byte address (low 32 bits
// of the generic shared pointer), gptr a 64-bit global address.
__device__ __forceinline__ void async_copy_b128(unsigned ldsaddr, const void* gptr) {
    asm volatile("global_load_async_to_lds_b128 %0, %1, off"
                 :: "v"(ldsaddr), "v"(gptr) : "memory");
}
__device__ __forceinline__ void async_wait0() {
    asm volatile("s_wait_asynccnt 0x0" ::: "memory");
}
__device__ __forceinline__ unsigned lds_addr32(const void* p) {
    return (unsigned)(uintptr_t)p;   // LDS aperture: addr[31:0] = LDS offset
}

// ---------------------------------------------------------------------------
// Elementwise: x = x + pos_embed (broadcast over batch), fp32
// ---------------------------------------------------------------------------
__global__ void add_pos_kernel(const float* __restrict__ x,
                               const float* __restrict__ pos,
                               float* __restrict__ out, long total, long nd)
{
    long i = (long)blockIdx.x * blockDim.x + threadIdx.x;
    if (i < total) out[i] = x[i] + pos[i % nd];
}

// ---------------------------------------------------------------------------
// fp32 -> f16 conversion (vectorized x4); n4 = count/4
// ---------------------------------------------------------------------------
__global__ void cvt_kernel(const float* __restrict__ src,
                           _Float16* __restrict__ dst, long n4)
{
    long i = (long)blockIdx.x * blockDim.x + threadIdx.x;
    if (i < n4) {
        v4f f = *(const v4f*)(src + i * 4);
        v4h h;
#pragma unroll
        for (int j = 0; j < 4; ++j) h[j] = (_Float16)f[j];
        *(v4h*)(dst + i * 4) = h;
    }
}

// ---------------------------------------------------------------------------
// LayerNorm over last dim (D=768). One block (256 threads) per row.
// OutT = _Float16 (feeds WMMA GEMMs) or float (final output).
// ---------------------------------------------------------------------------
template<typename OutT>
__global__ void ln_kernel(const float* __restrict__ x,
                          const float* __restrict__ w,
                          const float* __restrict__ b,
                          OutT* __restrict__ out)
{
    const int row = blockIdx.x;
    const float* xr = x + (size_t)row * 768;
    __shared__ float s1[256], s2[256];
    float a0 = 0.f, a1 = 0.f;
    for (int d = threadIdx.x; d < 768; d += 256) {
        float v = xr[d];
        a0 += v; a1 += v * v;
    }
    s1[threadIdx.x] = a0; s2[threadIdx.x] = a1;
    __syncthreads();
    for (int off = 128; off > 0; off >>= 1) {
        if (threadIdx.x < off) {
            s1[threadIdx.x] += s1[threadIdx.x + off];
            s2[threadIdx.x] += s2[threadIdx.x + off];
        }
        __syncthreads();
    }
    float mu  = s1[0] * (1.0f / 768.0f);
    float var = s2[0] * (1.0f / 768.0f) - mu * mu;
    float inv = rsqrtf(var + 1e-5f);
    for (int d = threadIdx.x; d < 768; d += 256)
        out[(size_t)row * 768 + d] = (OutT)((xr[d] - mu) * inv * w[d] + b[d]);
}

// ---------------------------------------------------------------------------
// GEMM: C[M,N] = A[M,K] @ W[N,K]^T (+bias) (+epilogue). A, W are f16.
// 128 threads = 4 waves; 64x64 output tile; WMMA f16 inputs, f32 accumulate.
// Tiles staged via GLOBAL_LOAD_ASYNC_TO_LDS_B128 (pure f16 copies).
// OP: 1 = exact GELU -> store f16, 2 = residual add -> store f32,
//     3 = store f16. (C reinterpreted as _Float16* for OP 1/3.)
// M, N, K are multiples of 64/32 here -> no bounds checks.
// ---------------------------------------------------------------------------
template<int OP>
__global__ void __launch_bounds__(128)
gemm_kernel(const _Float16* __restrict__ A,
            const _Float16* __restrict__ W,
            const float* __restrict__ bias,
            float* __restrict__ C,
            int M, int N, int K)
{
    __shared__ _Float16 sA[64][32];   // [m][k]
    __shared__ _Float16 sBt[64][32];  // [n][k]
    const int tid  = threadIdx.x;
    const int lane = tid & 31;
    const int wave = tid >> 5;        // 0..3
    const int m0 = blockIdx.x * 64;
    const int n0 = blockIdx.y * 64;
    const int lr = lane & 15;

    v8f acc[4];
#pragma unroll
    for (int s = 0; s < 4; ++s)
#pragma unroll
        for (int i = 0; i < 8; ++i) acc[s][i] = 0.0f;

    for (int k0 = 0; k0 < K; k0 += 32) {
        // stage A and W tiles: 64 rows x 32 halves = 4 b128 chunks per row
        for (int c = tid; c < 512; c += 128) {
            const int which = c >> 8;          // 0 = A, 1 = W
            const int r     = (c >> 2) & 63;
            const int part  = c & 3;
            const _Float16* src = which ? (W + (size_t)(n0 + r) * K + k0 + part * 8)
                                        : (A + (size_t)(m0 + r) * K + k0 + part * 8);
            unsigned dst = which ? lds_addr32(&sBt[r][part * 8])
                                 : lds_addr32(&sA [r][part * 8]);
            async_copy_b128(dst, src);
        }
        async_wait0();
        __syncthreads();

        v16h afrag;
        const int ar = wave * 16 + lr;
#pragma unroll
        for (int j = 0; j < 16; ++j) afrag[j] = sA[ar][frag_k(lane, j)];

#pragma unroll
        for (int s = 0; s < 4; ++s) {
            v16h bfrag;
            const int bc = s * 16 + lr;
#pragma unroll
            for (int j = 0; j < 16; ++j) bfrag[j] = sBt[bc][frag_k(lane, j)];
            acc[s] = __builtin_amdgcn_wmma_f32_16x16x32_f16(
                false, afrag, false, bfrag, (short)0, acc[s], false, false);
        }
        __syncthreads();
    }

#pragma unroll
    for (int s = 0; s < 4; ++s) {
        const int col = n0 + s * 16 + lr;
        const float bv = bias ? bias[col] : 0.0f;
#pragma unroll
        for (int i = 0; i < 8; ++i) {
            const int row = m0 + wave * 16 + ((lane >> 4) << 3) + i;
            float v = acc[s][i] + bv;
            size_t o = (size_t)row * N + col;
            if (OP == 1) {
                v = 0.5f * v * (1.0f + erff(v * 0.70710678118654752f));
                ((_Float16*)C)[o] = (_Float16)v;
            } else if (OP == 2) {
                C[o] = v + C[o];
            } else {
                ((_Float16*)C)[o] = (_Float16)v;
            }
        }
    }
}

// ---------------------------------------------------------------------------
// Attention: one workgroup (128 thr = 4 waves) per (batch b, 16-row tile t).
// Loops 12 heads sequentially (deterministic head sums). qkv/y are f16;
// qkv layout [B,392,2304], row = [q(768) | k(768) | v(768)]; y[b,n,h*64+d].
// Q/K/V tiles staged via async-to-LDS b128 copies; boundary rows zero-filled.
// Raw-score head sums accumulate in LDS (sAcc); out3 in registers.
// ---------------------------------------------------------------------------
__global__ void __launch_bounds__(128)
attn_kernel(const _Float16* __restrict__ qkv,
            _Float16* __restrict__ y,   // [B,392,768] f16
            float* __restrict__ out2,   // [B,391,391] (last layer only)
            float* __restrict__ out3,   // [B,391]     (last layer only)
            int lastLayer)
{
    const int t    = blockIdx.x;   // row tile 0..24
    const int b    = blockIdx.y;   // batch
    const int tid  = threadIdx.x;  // 0..127
    const int lane = tid & 31;
    const int wave = tid >> 5;
    const int lr   = lane & 15;
    const int khalf = lane >> 4;

    __shared__ float    sS[16][416];    // scores / probs
    __shared__ float    sAcc[16][392];  // head-sum of raw scores (last layer)
    __shared__ _Float16 sQ[16][64];
    __shared__ _Float16 sKV[64][64];

    const v8h zero8 = {};
    float o3[4] = {0.f, 0.f, 0.f, 0.f};

    if (lastLayer) {
        for (int idx = tid; idx < 16 * 392; idx += 128)
            (&sAcc[0][0])[idx] = 0.0f;
    }
    __syncthreads();

    for (int h = 0; h < 12; ++h) {
        const size_t base = (size_t)b * 392 * 2304 + (size_t)h * 64;

        // ---- stage Q tile: 16 rows x 64 halves (8 b128 chunks per row) ----
        {
            const int row = tid >> 3, part = tid & 7;      // 128 chunks
            const int n = t * 16 + row;
            if (n < 392)
                async_copy_b128(lds_addr32(&sQ[row][part * 8]),
                                qkv + base + (size_t)n * 2304 + part * 8);
            else
                *(v8h*)&sQ[row][part * 8] = zero8;
        }
        async_wait0();
        __syncthreads();

        // ---- phase 1: raw scores S = (Q @ K^T) * 0.125 into sS -------------
        for (int g = 0; g < 7; ++g) {                      // 7 groups of 4 col tiles
            const int m0 = g * 64;
            int validRows = 392 - m0; if (validRows > 64) validRows = 64;
            for (int c = tid; c < 512; c += 128) {
                const int row = c >> 3, part = c & 7;
                if (row < validRows)
                    async_copy_b128(lds_addr32(&sKV[row][part * 8]),
                                    qkv + base + (size_t)(m0 + row) * 2304 + 768 + part * 8);
                else
                    *(v8h*)&sKV[row][part * 8] = zero8;
            }
            async_wait0();
            __syncthreads();

            const int jt = g * 4 + wave;
            if (jt < 25) {
                v8f acc;
#pragma unroll
                for (int i = 0; i < 8; ++i) acc[i] = 0.0f;
#pragma unroll
                for (int ks = 0; ks < 2; ++ks) {
                    const int k0 = ks * 32;
                    v16h afrag, bfrag;
#pragma unroll
                    for (int j = 0; j < 16; ++j) afrag[j] = sQ[lr][k0 + frag_k(lane, j)];
#pragma unroll
                    for (int j = 0; j < 16; ++j) bfrag[j] = sKV[wave * 16 + lr][k0 + frag_k(lane, j)];
                    acc = __builtin_amdgcn_wmma_f32_16x16x32_f16(
                        false, afrag, false, bfrag, (short)0, acc, false, false);
                }
#pragma unroll
                for (int i = 0; i < 8; ++i)
                    sS[khalf * 8 + i][jt * 16 + lr] = acc[i] * 0.125f;
            }
            __syncthreads();
        }

        // ---- phase 2: accumulate raw scores into LDS head-sum -------------
        if (lastLayer) {
            for (int idx = tid; idx < 16 * 392; idx += 128) {
                int r = idx / 392, m = idx % 392;
                sAcc[r][m] += sS[r][m];
            }
            __syncthreads();
        }

        // ---- phase 3: softmax each row; zero padded columns ---------------
        if (tid < 16) {
            const int r = tid;
            const int n = t * 16 + r;
            if (n < 392) {
                float mx = -3.4e38f;
                for (int c = 0; c < 392; ++c) mx = fmaxf(mx, sS[r][c]);
                float sum = 0.f;
                for (int c = 0; c < 392; ++c) {
                    float e = expf(sS[r][c] - mx);
                    sS[r][c] = e; sum += e;
                }
                float inv = 1.0f / sum;
                for (int c = 0; c < 392; ++c) sS[r][c] *= inv;
                for (int c = 392; c < 416; ++c) sS[r][c] = 0.0f;
            } else {
                for (int c = 0; c < 416; ++c) sS[r][c] = 0.0f;
            }
        }
        __syncthreads();

        // ---- phase 4: accumulate softmax row 0 into registers (out3) ------
        if (lastLayer && t == 0) {
#pragma unroll
            for (int q = 0; q < 4; ++q) {
                int m = 1 + tid + q * 128;
                if (m < 392) o3[q] += sS[0][m];
            }
        }

        // ---- phase 5: y = P @ V; wave w owns d-subtile w (16 of 64) -------
        {
            v8f acc;
#pragma unroll
            for (int i = 0; i < 8; ++i) acc[i] = 0.0f;
            for (int kt = 0; kt < 13; ++kt) {
                const int m0 = kt * 32;
                int validRows = 392 - m0;
                if (validRows > 32) validRows = 32;
                if (validRows < 0)  validRows = 0;
                for (int c = tid; c < 256; c += 128) {
                    const int row = c >> 3, part = c & 7;
                    if (row < validRows)
                        async_copy_b128(lds_addr32(&sKV[row][part * 8]),
                                        qkv + base + (size_t)(m0 + row) * 2304 + 1536 + part * 8);
                    else
                        *(v8h*)&sKV[row][part * 8] = zero8;
                }
                async_wait0();
                __syncthreads();

                v16h afrag, bfrag;
#pragma unroll
                for (int j = 0; j < 16; ++j)
                    afrag[j] = (_Float16)sS[lr][m0 + frag_k(lane, j)];
#pragma unroll
                for (int j = 0; j < 16; ++j)
                    bfrag[j] = sKV[frag_k(lane, j)][wave * 16 + lr];
                acc = __builtin_amdgcn_wmma_f32_16x16x32_f16(
                    false, afrag, false, bfrag, (short)0, acc, false, false);
                __syncthreads();
            }
#pragma unroll
            for (int i = 0; i < 8; ++i) {
                int r = khalf * 8 + i;
                int n = t * 16 + r;
                if (n < 392)
                    y[((size_t)b * 392 + n) * 768 + (size_t)h * 64 + wave * 16 + lr]
                        = (_Float16)acc[i];
            }
        }
        __syncthreads();
    }

    // ---- write accumulated outputs (last layer) ---------------------------
    if (lastLayer) {
        for (int idx = tid; idx < 16 * 391; idx += 128) {
            int r = idx / 391;
            int m = 1 + idx % 391;
            int n = t * 16 + r;
            if (n >= 1 && n < 392)
                out2[((size_t)b * 391 + (n - 1)) * 391 + (m - 1)] = sAcc[r][m];
        }
        if (t == 0) {
#pragma unroll
            for (int q = 0; q < 4; ++q) {
                int m = 1 + tid + q * 128;
                if (m < 392) out3[(size_t)b * 391 + (m - 1)] = o3[q];
            }
        }
    }
}

// ---------------------------------------------------------------------------
// Host launcher
// ---------------------------------------------------------------------------
extern "C" void kernel_launch(void* const* d_in, const int* in_sizes, int n_in,
                              void* d_out, int out_size, void* d_ws, size_t ws_size,
                              hipStream_t stream)
{
    const float* x_in   = (const float*)d_in[0];
    const float* pos    = (const float*)d_in[1];
    const float* ln1_w  = (const float*)d_in[2];
    const float* ln1_b  = (const float*)d_in[3];
    const float* qkv_w  = (const float*)d_in[4];
    const float* proj_w = (const float*)d_in[5];
    const float* proj_b = (const float*)d_in[6];
    const float* ln2_w  = (const float*)d_in[7];
    const float* ln2_b  = (const float*)d_in[8];
    const float* fc1_w  = (const float*)d_in[9];
    const float* fc1_b  = (const float*)d_in[10];
    const float* fc2_w  = (const float*)d_in[11];
    const float* fc2_b  = (const float*)d_in[12];
    const float* norm_w = (const float*)d_in[13];
    const float* norm_b = (const float*)d_in[14];

    const int B = 16, N = 392, D = 768, FF = 3072, L = 12;
    const size_t MR = (size_t)B * N;        // 6272 rows

    float* out  = (float*)d_out;
    float* out1 = out;                                        // [16,392,768]
    float* out2 = out1 + (size_t)B * N * D;                   // [16,391,391]
    float* out3 = out2 + (size_t)B * (N - 1) * (N - 1);       // [16,391]

    // workspace (float units; f16 regions take half the floats)
    float* ws = (float*)d_ws;
    float*    xbuf   = ws;                               // MR*768 f32
    _Float16* hbuf   = (_Float16*)(xbuf + MR * D);       // MR*768 f16
    _Float16* qkv_h  = (_Float16*)((float*)hbuf + MR * D / 2);       // MR*2304 f16
    _Float16* ybuf   = (_Float16*)((float*)qkv_h + MR * 3 * D / 2);  // MR*768 f16
    _Float16* mlp_h  = (_Float16*)((float*)ybuf + MR * D / 2);       // MR*3072 f16
    _Float16* wqkv_h = (_Float16*)((float*)mlp_h + MR * FF / 2);     // 2304*768 f16
    _Float16* wproj_h= wqkv_h + (size_t)3 * D * D;
    _Float16* wfc1_h = wproj_h + (size_t)D * D;
    _Float16* wfc2_h = wfc1_h + (size_t)FF * D;

    // x = x + pos_embed
    {
        long total = (long)MR * D;
        long nd = (long)N * D;
        add_pos_kernel<<<dim3((unsigned)((total + 255) / 256)), 256, 0, stream>>>(
            x_in, pos, xbuf, total, nd);
    }

    for (int l = 0; l < L; ++l) {
        // convert this layer's weights to f16
        {
            long n4;
            n4 = (long)3 * D * D / 4;
            cvt_kernel<<<dim3((unsigned)((n4 + 255) / 256)), 256, 0, stream>>>(
                qkv_w + (size_t)l * 3 * D * D, wqkv_h, n4);
            n4 = (long)D * D / 4;
            cvt_kernel<<<dim3((unsigned)((n4 + 255) / 256)), 256, 0, stream>>>(
                proj_w + (size_t)l * D * D, wproj_h, n4);
            n4 = (long)FF * D / 4;
            cvt_kernel<<<dim3((unsigned)((n4 + 255) / 256)), 256, 0, stream>>>(
                fc1_w + (size_t)l * FF * D, wfc1_h, n4);
            cvt_kernel<<<dim3((unsigned)((n4 + 255) / 256)), 256, 0, stream>>>(
                fc2_w + (size_t)l * D * FF, wfc2_h, n4);
        }
        // LN1 -> f16
        ln_kernel<_Float16><<<dim3((unsigned)MR), 256, 0, stream>>>(
            xbuf, ln1_w + (size_t)l * D, ln1_b + (size_t)l * D, hbuf);
        // QKV: [6272,2304] f16 = h @ qkv_w^T (no bias)
        gemm_kernel<3><<<dim3(98, 3 * D / 64), 128, 0, stream>>>(
            hbuf, wqkv_h, nullptr, (float*)qkv_h, (int)MR, 3 * D, D);
        // Attention (+ out2/out3 on last layer)
        attn_kernel<<<dim3(25, B), 128, 0, stream>>>(
            qkv_h, ybuf, out2, out3, (l == L - 1) ? 1 : 0);
        // proj + residual: x += y @ proj_w^T + proj_b  (f32 out)
        gemm_kernel<2><<<dim3(98, D / 64), 128, 0, stream>>>(
            ybuf, wproj_h, proj_b + (size_t)l * D, xbuf, (int)MR, D, D);
        // LN2 -> f16
        ln_kernel<_Float16><<<dim3((unsigned)MR), 256, 0, stream>>>(
            xbuf, ln2_w + (size_t)l * D, ln2_b + (size_t)l * D, hbuf);
        // fc1 + exact GELU -> f16
        gemm_kernel<1><<<dim3(98, FF / 64), 128, 0, stream>>>(
            hbuf, wfc1_h, fc1_b + (size_t)l * FF, (float*)mlp_h, (int)MR, FF, D);
        // fc2 + residual -> f32
        gemm_kernel<2><<<dim3(98, D / 64), 128, 0, stream>>>(
            mlp_h, wfc2_h, fc2_b + (size_t)l * D, xbuf, (int)MR, D, FF);
    }

    // final LN -> out1 (f32)
    ln_kernel<float><<<dim3((unsigned)MR), 256, 0, stream>>>(xbuf, norm_w, norm_b, out1);
}
